// EdgeGATConv_36584531427426
// MI455X (gfx1250) — matrix operand order repaired and checked by reading, not compiled, not run on hardware
//
#include <hip/hip_runtime.h>
#include <hip/hip_bf16.h>
#include <math.h>

// ---------------------------------------------------------------------------
// EdgeGATConv, fused for MI455X (gfx1250, wave32, WMMA).
//   H=4, F=16 -> H*F=64.  NODE_F=128, EDGE_F=64.
// Projections collapsed through attn vectors:
//   el/er = nfeat @ Wl|Wr  ([128 x 8] padded to 16 cols, f16 WMMA, f32 acc)
//   ee    = efeat @ We     ([64  x 4] padded to 16 cols, f16 WMMA, f32 acc)
// Weights are pre-folded *directly into the per-lane WMMA B-fragment layout*
// (f16), so projection waves load B with one contiguous 32B vector load.
// Then 3 edge passes for segment softmax + scatter-sum.
// ---------------------------------------------------------------------------

typedef __attribute__((ext_vector_type(16))) _Float16 v16h;
typedef __attribute__((ext_vector_type(8)))  float    v8f;

#define NEG_SLOPE 0.2f

// ---------------- weight folding into B fragments ---------------------------
// B-fragment layout (16-bit B, 32x16 KxN, wave32):
//   lane L holds column n = L&15;  halfs i=0..15 hold K = (L>>4)*8 + (i&7) + (i>>3)*16
//   within a 32-deep K block.  Frag[kblock][lane][i], 16 halfs per lane.
// Node: 4 kblocks (K=128), cols 0-3 = W_node.attn_l, 4-7 = W_node.attn_r.
// Edge: 2 kblocks (K=64),  cols 0-3 = W_edge.attn_e.
__global__ void gat_fold_weights(const float* __restrict__ W_node,
                                 const float* __restrict__ b_node,
                                 const float* __restrict__ W_edge,
                                 const float* __restrict__ b_edge,
                                 const float* __restrict__ attn_l,
                                 const float* __restrict__ attn_r,
                                 const float* __restrict__ attn_e,
                                 _Float16* __restrict__ BfragN,   // [4*32*16]
                                 _Float16* __restrict__ BfragE,   // [2*32*16]
                                 float* __restrict__ bcn,         // [16]
                                 float* __restrict__ bce) {       // [16]
    int t = threadIdx.x;
    // node fragments: 4 kblocks * 32 lanes * 16 halfs = 2048 entries
    for (int idx = t; idx < 2048; idx += blockDim.x) {
        int kblock = idx >> 9;
        int lane   = (idx >> 4) & 31;
        int i      = idx & 15;
        int K      = kblock * 32 + (lane >> 4) * 8 + (i & 7) + (i >> 3) * 16;  // 0..127
        int col    = lane & 15;
        float v = 0.f;
        if (col < 8) {
            int h = col & 3;
            const float* attn = (col < 4) ? attn_l : attn_r;
            #pragma unroll
            for (int f = 0; f < 16; ++f)
                v += W_node[K * 64 + h * 16 + f] * attn[h * 16 + f];
        }
        BfragN[idx] = (_Float16)v;
    }
    // edge fragments: 2 kblocks * 32 lanes * 16 halfs = 1024 entries
    for (int idx = t; idx < 1024; idx += blockDim.x) {
        int kblock = idx >> 9;
        int lane   = (idx >> 4) & 31;
        int i      = idx & 15;
        int K      = kblock * 32 + (lane >> 4) * 8 + (i & 7) + (i >> 3) * 16;  // 0..63
        int col    = lane & 15;
        float v = 0.f;
        if (col < 4) {
            #pragma unroll
            for (int f = 0; f < 16; ++f)
                v += W_edge[K * 64 + col * 16 + f] * attn_e[col * 16 + f];
        }
        BfragE[idx] = (_Float16)v;
    }
    if (t < 16) {
        float v = 0.f;
        if (t < 8) {
            int h = t & 3;
            const float* attn = (t < 4) ? attn_l : attn_r;
            #pragma unroll
            for (int f = 0; f < 16; ++f)
                v += b_node[h * 16 + f] * attn[h * 16 + f];
        }
        bcn[t] = v;
        float w = 0.f;
        if (t < 4) {
            #pragma unroll
            for (int f = 0; f < 16; ++f)
                w += b_edge[t * 16 + f] * attn_e[t * 16 + f];
        }
        bce[t] = w;
    }
}

__global__ void gat_fill(float* __restrict__ p, float v, int n) {
    for (int i = blockIdx.x * blockDim.x + threadIdx.x; i < n;
         i += blockDim.x * gridDim.x)
        p[i] = v;
}

// ---- A-fragment loader: 16-bit A 16x32 (MxK), lane half selects K groups ---
// lanes 0-15 (M=lane):  halfs 0-7 -> K=0..7,  halfs 8-15 -> K=16..23
// lanes 16-31 (M=l-16): halfs 0-7 -> K=8..15, halfs 8-15 -> K=24..31
__device__ __forceinline__ v16h gat_load_afrag(const float* __restrict__ arow) {
    // arow already offset by kb + half_id*8; contiguous chunks at +0 and +16
    float4 p0 = *(const float4*)(arow + 0);
    float4 p1 = *(const float4*)(arow + 4);
    float4 p2 = *(const float4*)(arow + 16);
    float4 p3 = *(const float4*)(arow + 20);
    v16h a;
    a[0]  = (_Float16)p0.x; a[1]  = (_Float16)p0.y;
    a[2]  = (_Float16)p0.z; a[3]  = (_Float16)p0.w;
    a[4]  = (_Float16)p1.x; a[5]  = (_Float16)p1.y;
    a[6]  = (_Float16)p1.z; a[7]  = (_Float16)p1.w;
    a[8]  = (_Float16)p2.x; a[9]  = (_Float16)p2.y;
    a[10] = (_Float16)p2.z; a[11] = (_Float16)p2.w;
    a[12] = (_Float16)p3.x; a[13] = (_Float16)p3.y;
    a[14] = (_Float16)p3.z; a[15] = (_Float16)p3.w;
    return a;
}

// ---------------- WMMA node projection -------------------------------------
// One wave per 16-node tile. K = 128 -> four 16x16x32 f16 WMMAs, f32 acc.
__global__ void gat_node_proj_wmma(const float* __restrict__ nfeat,
                                   const _Float16* __restrict__ BfragN,
                                   const float* __restrict__ bcn,
                                   float* __restrict__ el,   // [N*4]
                                   float* __restrict__ er,   // [N*4]
                                   int Nn, int ntiles) {
    int wave = (blockIdx.x * blockDim.x + threadIdx.x) >> 5;
    if (wave >= ntiles) return;                       // wave-uniform exit
    int lane    = threadIdx.x & 31;
    int half_id = lane >> 4;
    int mn      = lane & 15;
    int row0    = wave * 16;

    int r = row0 + mn; if (r >= Nn) r = Nn - 1;       // clamp (safe dup read)
    const float* arow = nfeat + (size_t)r * 128 + half_id * 8;
    __builtin_prefetch(nfeat + (size_t)(row0 + 16) * 128, 0, 0);

    v8f c = {};
    #pragma unroll
    for (int kb = 0; kb < 4; ++kb) {
        v16h a = gat_load_afrag(arow + kb * 32);
        v16h b = *(const v16h*)(BfragN + (size_t)kb * 512 + (size_t)lane * 16);
        c = __builtin_amdgcn_wmma_f32_16x16x32_f16(false, a, false, b,
                                                   (short)0, c, false, false);
    }
    // C layout: VGPR i holds M = i + 8*half_id, N = lane&15
    float bias = bcn[mn];
    if (mn < 8) {
        float* dstp = (mn < 4) ? el : er;
        int    col  = mn & 3;
        #pragma unroll
        for (int i = 0; i < 8; ++i) {
            int row = row0 + i + 8 * half_id;
            if (row < Nn)
                dstp[(size_t)row * 4 + col] = c[i] + bias;
        }
    }
}

// ---------------- WMMA edge projection -------------------------------------
// One wave per 16-edge tile. K = 64 -> two 16x16x32 f16 WMMAs.
__global__ void gat_edge_proj_wmma(const float* __restrict__ efeat,
                                   const _Float16* __restrict__ BfragE,
                                   const float* __restrict__ bce,
                                   float* __restrict__ ee,   // [E*4]
                                   int Ee, int ntiles) {
    int wave = (blockIdx.x * blockDim.x + threadIdx.x) >> 5;
    if (wave >= ntiles) return;                       // wave-uniform exit
    int lane    = threadIdx.x & 31;
    int half_id = lane >> 4;
    int mn      = lane & 15;
    int row0    = wave * 16;

    int r = row0 + mn; if (r >= Ee) r = Ee - 1;
    const float* arow = efeat + (size_t)r * 64 + half_id * 8;
    __builtin_prefetch(efeat + (size_t)(row0 + 16) * 64, 0, 0);

    v8f c = {};
    #pragma unroll
    for (int kb = 0; kb < 2; ++kb) {
        v16h a = gat_load_afrag(arow + kb * 32);
        v16h b = *(const v16h*)(BfragE + (size_t)kb * 512 + (size_t)lane * 16);
        c = __builtin_amdgcn_wmma_f32_16x16x32_f16(false, a, false, b,
                                                   (short)0, c, false, false);
    }
    float bias = bce[mn];
    if (mn < 4) {
        #pragma unroll
        for (int i = 0; i < 8; ++i) {
            int row = row0 + i + 8 * half_id;
            if (row < Ee)
                ee[(size_t)row * 4 + mn] = c[i] + bias;
        }
    }
}

// ---------------- edge softmax passes ---------------------------------------
__device__ __forceinline__ void atomicMaxF(float* addr, float val) {
    int* ai = reinterpret_cast<int*>(addr);
    int old = __float_as_int(*addr);
    while (__int_as_float(old) < val) {
        int assumed = old;
        old = atomicCAS(ai, assumed, __float_as_int(val));
        if (old == assumed) break;
    }
}

__device__ __forceinline__ float4 gat_edge_e(const float4* el4, const float4* er4,
                                             const float4* ee4, int s, int d,
                                             int e, float4* elp_out) {
    float4 elv = el4[s], erv = er4[d], eev = ee4[e];
    float4 elp = make_float4(elv.x + eev.x, elv.y + eev.y,
                             elv.z + eev.z, elv.w + eev.w);
    float4 t = make_float4(elp.x + erv.x, elp.y + erv.y,
                           elp.z + erv.z, elp.w + erv.w);
    t.x = t.x > 0.f ? t.x : NEG_SLOPE * t.x;
    t.y = t.y > 0.f ? t.y : NEG_SLOPE * t.y;
    t.z = t.z > 0.f ? t.z : NEG_SLOPE * t.z;
    t.w = t.w > 0.f ? t.w : NEG_SLOPE * t.w;
    *elp_out = elp;
    return t;
}

__global__ void gat_pass_max(const float* __restrict__ el, const float* __restrict__ er,
                             const float* __restrict__ ee, const int* __restrict__ src,
                             const int* __restrict__ dst, float* __restrict__ emax, int Ee) {
    int e = blockIdx.x * blockDim.x + threadIdx.x;
    if (e >= Ee) return;
    int s = src[e], d = dst[e];
    float4 elp;
    float4 t = gat_edge_e((const float4*)el, (const float4*)er,
                          (const float4*)ee, s, d, e, &elp);
    atomicMaxF(&emax[(size_t)d * 4 + 0], t.x);
    atomicMaxF(&emax[(size_t)d * 4 + 1], t.y);
    atomicMaxF(&emax[(size_t)d * 4 + 2], t.z);
    atomicMaxF(&emax[(size_t)d * 4 + 3], t.w);
}

__global__ void gat_pass_sum(const float* __restrict__ el, const float* __restrict__ er,
                             const float* __restrict__ ee, const int* __restrict__ src,
                             const int* __restrict__ dst, const float* __restrict__ emax,
                             float* __restrict__ denom, int Ee) {
    int e = blockIdx.x * blockDim.x + threadIdx.x;
    if (e >= Ee) return;
    int s = src[e], d = dst[e];
    float4 elp;
    float4 t = gat_edge_e((const float4*)el, (const float4*)er,
                          (const float4*)ee, s, d, e, &elp);
    float4 mx = ((const float4*)emax)[d];
    atomicAdd(&denom[(size_t)d * 4 + 0], __expf(t.x - mx.x));
    atomicAdd(&denom[(size_t)d * 4 + 1], __expf(t.y - mx.y));
    atomicAdd(&denom[(size_t)d * 4 + 2], __expf(t.z - mx.z));
    atomicAdd(&denom[(size_t)d * 4 + 3], __expf(t.w - mx.w));
}

__global__ void gat_pass_out(const float* __restrict__ el, const float* __restrict__ er,
                             const float* __restrict__ ee, const int* __restrict__ src,
                             const int* __restrict__ dst, const float* __restrict__ emax,
                             const float* __restrict__ denom, float* __restrict__ out, int Ee) {
    int e = blockIdx.x * blockDim.x + threadIdx.x;
    if (e >= Ee) return;
    int s = src[e], d = dst[e];
    float4 elp;
    float4 t = gat_edge_e((const float4*)el, (const float4*)er,
                          (const float4*)ee, s, d, e, &elp);
    float4 mx = ((const float4*)emax)[d];
    float4 dn = ((const float4*)denom)[d];
    atomicAdd(&out[(size_t)d * 4 + 0], (__expf(t.x - mx.x) / dn.x) * elp.x);
    atomicAdd(&out[(size_t)d * 4 + 1], (__expf(t.y - mx.y) / dn.y) * elp.y);
    atomicAdd(&out[(size_t)d * 4 + 2], (__expf(t.z - mx.z) / dn.z) * elp.z);
    atomicAdd(&out[(size_t)d * 4 + 3], (__expf(t.w - mx.w) / dn.w) * elp.w);
}

// ---------------------------------------------------------------------------
extern "C" void kernel_launch(void* const* d_in, const int* in_sizes, int n_in,
                              void* d_out, int out_size, void* d_ws, size_t ws_size,
                              hipStream_t stream) {
    const float* nfeat  = (const float*)d_in[0];
    const float* efeat  = (const float*)d_in[1];
    const float* W_node = (const float*)d_in[2];
    const float* b_node = (const float*)d_in[3];
    const float* W_edge = (const float*)d_in[4];
    const float* b_edge = (const float*)d_in[5];
    const float* attn_l = (const float*)d_in[6];
    const float* attn_r = (const float*)d_in[7];
    const float* attn_e = (const float*)d_in[8];
    const int*   src    = (const int*)d_in[9];
    const int*   dst    = (const int*)d_in[10];
    float*       out    = (float*)d_out;

    const int Nn = in_sizes[0] / 128;   // nodes
    const int Ee = in_sizes[9];         // edges

    // ---- workspace carve-up ----
    float* bcn   = (float*)d_ws;              // 16
    float* bce   = bcn + 16;                  // 16
    float* el    = bce + 16;                  // Nn*4  (offset 128B, 16B align)
    float* er    = el  + (size_t)Nn * 4;      // Nn*4
    float* ee    = er  + (size_t)Nn * 4;      // Ee*4
    float* emax  = ee  + (size_t)Ee * 4;      // Nn*4
    float* denom = emax + (size_t)Nn * 4;     // Nn*4
    uintptr_t fp = (uintptr_t)(denom + (size_t)Nn * 4);
    fp = (fp + 63) & ~(uintptr_t)63;          // 64B align fragment buffers
    _Float16* BfragN = (_Float16*)fp;         // 4*32*16 halfs (4 KB)
    _Float16* BfragE = BfragN + 2048;         // 2*32*16 halfs (2 KB)

    // 1) fold weights through attention vectors, directly into B fragments
    gat_fold_weights<<<1, 256, 0, stream>>>(W_node, b_node, W_edge, b_edge,
                                            attn_l, attn_r, attn_e,
                                            BfragN, BfragE, bcn, bce);

    // 2) init reductions
    int nh = Nn * 4;
    gat_fill<<<(nh + 255) / 256, 256, 0, stream>>>(emax, -INFINITY, nh);
    gat_fill<<<(nh + 255) / 256, 256, 0, stream>>>(denom, 0.f, nh);
    gat_fill<<<(out_size + 255) / 256, 256, 0, stream>>>(out, 0.f, out_size);

    // 3) WMMA projections (8 waves / 256-thread block, 1 tile per wave)
    int ntile_n = (Nn + 15) / 16;
    gat_node_proj_wmma<<<(ntile_n + 7) / 8, 256, 0, stream>>>(
        nfeat, BfragN, bcn, el, er, Nn, ntile_n);
    int ntile_e = (Ee + 15) / 16;
    gat_edge_proj_wmma<<<(ntile_e + 7) / 8, 256, 0, stream>>>(
        efeat, BfragE, bce, ee, Ee, ntile_e);

    // 4) segment softmax + scatter-sum (3 passes over edges)
    int eb = (Ee + 255) / 256;
    gat_pass_max<<<eb, 256, 0, stream>>>(el, er, ee, src, dst, emax, Ee);
    gat_pass_sum<<<eb, 256, 0, stream>>>(el, er, ee, src, dst, emax, denom, Ee);
    gat_pass_out<<<eb, 256, 0, stream>>>(el, er, ee, src, dst, emax, denom, out, Ee);
}
